// LogicVAE_52012053954609
// MI455X (gfx1250) — matrix operand order repaired and checked by reading, not compiled, not run on hardware
//
#include <hip/hip_runtime.h>
#include <hip/hip_bf16.h>

#define Bsz 2048
#define Nn  32
#define Vv  10
#define Hh  256
#define Zz  64
#define TH3 768
#define MR  32            // batch rows per workgroup
#define NWAVE 16          // 512 threads

typedef __attribute__((ext_vector_type(16))) __bf16 v16bf;
typedef __attribute__((ext_vector_type(8)))  __bf16 v8bf;
typedef __attribute__((ext_vector_type(8)))  float  v8f;

union BF16x16 { v16bf v; v8bf h[2]; };

__device__ __forceinline__ v16bf cat16(v8bf lo, v8bf hi) {
  BF16x16 u; u.h[0] = lo; u.h[1] = hi; return u.v;
}

// A fragment (16x32 bf16) from row-major 16xH LDS tile.
// Lane l: row = l%16 ; VGPR0-3 hold K = kc*32 + (l/16)*8 .. +7 ; VGPR4-7 hold +16.
__device__ __forceinline__ v16bf load_A_lds(const __bf16* s, int kc, int lane) {
  const __bf16* p = s + (lane & 15) * Hh + (kc << 5) + ((lane >> 4) << 3);
  return cat16(*(const v8bf*)p, *(const v8bf*)(p + 16));
}

// B fragment (32x16 bf16) from pre-packed weights: [(tile*8+kc)*32 + lane][16] contiguous.
__device__ __forceinline__ v16bf load_B_pk(const __bf16* pk, int t, int kc, int lane) {
  const __bf16* p = pk + (size_t)((((t << 3) + kc) << 5) + lane) * 16;
  return cat16(*(const v8bf*)p, *(const v8bf*)(p + 8));
}

__device__ __forceinline__ v8f wmma_bf(v16bf a, v16bf b, v8f c) {
  return __builtin_amdgcn_wmma_f32_16x16x32_bf16(false, a, false, b, (short)0, c, false, false);
}

__device__ __forceinline__ v8f vzero() {
  v8f z;
#pragma unroll
  for (int e = 0; e < 8; ++e) z[e] = 0.0f;
  return z;
}

__device__ __forceinline__ float sigm(float x) { return 1.0f / (1.0f + __expf(-x)); }

// Pack f32 (K x Ncols, row-major) weights into bf16 B-fragment order:
// p -> tile t, kc, lane l, e : n = t*16 + l%16, k = kc*32 + (l/16)*8 + (e%8) + (e/8)*16
__global__ void pack_w(const float* __restrict__ src, __bf16* __restrict__ dst, int K, int Ncols) {
  int p = blockIdx.x * blockDim.x + threadIdx.x;
  int kchunks = K >> 5;
  int total = (Ncols >> 4) * kchunks * 512;
  if (p >= total) return;
  int e  = p & 15;
  int l  = (p >> 4) & 31;
  int q  = p >> 9;
  int kc = q % kchunks;
  int t  = q / kchunks;
  int n  = (t << 4) + (l & 15);
  int k  = (kc << 5) + ((l >> 4) << 3) + (e & 7) + ((e >> 3) << 4);
  dst[p] = (__bf16)src[k * Ncols + n];
}

// Whole 32-step scan in ONE launch: each WG owns 32 batch rows end-to-end.
__global__ __launch_bounds__(512)
void lvae_scan(const float* __restrict__ adj, const int* __restrict__ ntypes,
               const float* __restrict__ w_ih, const float* __restrict__ b_ih,
               const float* __restrict__ b_hh, const float* __restrict__ b_gate,
               const float* __restrict__ b_mu, const float* __restrict__ b_std,
               const __bf16* __restrict__ whh_p, const __bf16* __restrict__ wg_p,
               const __bf16* __restrict__ wm_p, const __bf16* __restrict__ wmu_p,
               const __bf16* __restrict__ wsd_p,
               __bf16* __restrict__ gm, float* __restrict__ out)
{
  __shared__ __bf16 s_aggb[MR * Hh];  // bf16 agg (WMMA A + GRU h-prev)
  __shared__ __bf16 s_hnb[MR * Hh];   // bf16 h_new (WMMA A)
  __shared__ float  s_adj[MR * Nn];
  __shared__ int    s_type[MR];

  const int tid  = threadIdx.x;
  const int lane = tid & 31;
  const int w    = tid >> 5;          // wave 0..15
  const int b0   = blockIdx.x * MR;

#pragma unroll 1
  for (int v = 0; v < Nn; ++v) {
    // Launder a zero SGPR offset per step: addresses stay addrspace(1)
    // (global_load_b128, scalar base) but are no longer loop-invariant, so
    // B-fragment loads are re-issued from L2 each step instead of
    // hoisted-and-spilled across the whole scan.
    size_t lz = 0;
    asm volatile("" : "+s"(lz));
    const __bf16* whh_l = whh_p + lz;
    const __bf16* wg_l  = wg_p  + lz;
    const __bf16* wm_l  = wm_p  + lz;

    // ---- preload adjacency column v (strictly n < v) and node types ----
    for (int i = tid; i < MR * Nn; i += 512) {
      int r = i >> 5, n = i & 31;
      s_adj[i] = (n < v) ? adj[((size_t)(b0 + r) * Nn + n) * Nn + v] : 0.0f;
    }
    if (tid < MR) s_type[tid] = ntypes[(b0 + tid) * Nn + v];
    __syncthreads();  // also fences the previous step's gm stores for phase A

    // ---- Phase A: agg[r,h] = sum_{n<v} adj * gm (sparse weighted sum) ----
    for (int idx = tid; idx < MR * 32; idx += 512) {
      int r = idx >> 5;
      int h = (idx & 31) << 3;
      float acc[8];
#pragma unroll
      for (int e = 0; e < 8; ++e) acc[e] = 0.0f;
      const __bf16* gb = gm + (size_t)(b0 + r) * Nn * Hh + h;
      for (int n = 0; n < v; ++n) {
        float a = s_adj[(r << 5) + n];
        if (a != 0.0f) {
          v8bf g = *(const v8bf*)(gb + n * Hh);
#pragma unroll
          for (int e = 0; e < 8; ++e) acc[e] += a * (float)g[e];
        }
      }
#pragma unroll
      for (int e = 0; e < 8; ++e) s_aggb[r * Hh + h + e] = (__bf16)acc[e];
    }
    __syncthreads();

    // ---- Phase B: gh = agg @ w_hh. Wave w owns columns w*16..w*16+15 for all
    //      three gates (tiles w, w+16, w+32); each B frag feeds 2 M-subtiles. ----
    v8f accR[2], accZ[2], accN[2];
#pragma unroll
    for (int s = 0; s < 2; ++s) { accR[s] = vzero(); accZ[s] = vzero(); accN[s] = vzero(); }
#pragma unroll
    for (int kc = 0; kc < 8; ++kc) {
      v16bf a0 = load_A_lds(s_aggb,           kc, lane);   // rows 0..15
      v16bf a1 = load_A_lds(s_aggb + 16 * Hh, kc, lane);   // rows 16..31
      v16bf bR = load_B_pk(whh_l, w,      kc, lane);
      v16bf bZ = load_B_pk(whh_l, w + 16, kc, lane);
      v16bf bN = load_B_pk(whh_l, w + 32, kc, lane);
      accR[0] = wmma_bf(a0, bR, accR[0]);  accR[1] = wmma_bf(a1, bR, accR[1]);
      accZ[0] = wmma_bf(a0, bZ, accZ[0]);  accZ[1] = wmma_bf(a1, bZ, accZ[1]);
      accN[0] = wmma_bf(a0, bN, accN[0]);  accN[1] = wmma_bf(a1, bN, accN[1]);
    }

    // ---- Phase C: GRU combine in registers; gi = w_ih[type] row gather ----
    {
      int c = (w << 4) + (lane & 15);   // h column 0..255
#pragma unroll
      for (int s = 0; s < 2; ++s) {
#pragma unroll
        for (int r = 0; r < 8; ++r) {
          int m  = (s << 4) + r + ((lane >> 4) << 3);      // row 0..31
          int tt = s_type[m];
          float ir  = w_ih[tt * TH3 + c]       + b_ih[c];
          float iz  = w_ih[tt * TH3 + 256 + c] + b_ih[256 + c];
          float in_ = w_ih[tt * TH3 + 512 + c] + b_ih[512 + c];
          float hr = accR[s][r] + b_hh[c];
          float hz = accZ[s][r] + b_hh[256 + c];
          float hn = accN[s][r] + b_hh[512 + c];
          float rg = sigm(ir + hr);
          float zg = sigm(iz + hz);
          float ng = tanhf(in_ + rg * hn);
          float hp = (float)s_aggb[m * Hh + c];
          float hv = (1.0f - zg) * ng + zg * hp;
          s_hnb[m * Hh + c] = (__bf16)hv;
        }
      }
    }
    __syncthreads();

    if (v < Nn - 1) {
      // ---- Phase D: gm_v = sigmoid(h@w_gate+b)*(h@w_map); tile w of each ----
      v8f aG[2], aM[2];
#pragma unroll
      for (int s = 0; s < 2; ++s) { aG[s] = vzero(); aM[s] = vzero(); }
#pragma unroll
      for (int kc = 0; kc < 8; ++kc) {
        v16bf a0 = load_A_lds(s_hnb,           kc, lane);
        v16bf a1 = load_A_lds(s_hnb + 16 * Hh, kc, lane);
        v16bf bG = load_B_pk(wg_l, w, kc, lane);
        v16bf bM = load_B_pk(wm_l, w, kc, lane);
        aG[0] = wmma_bf(a0, bG, aG[0]);  aG[1] = wmma_bf(a1, bG, aG[1]);
        aM[0] = wmma_bf(a0, bM, aM[0]);  aM[1] = wmma_bf(a1, bM, aM[1]);
      }
      int c = (w << 4) + (lane & 15);
#pragma unroll
      for (int s = 0; s < 2; ++s) {
#pragma unroll
        for (int r = 0; r < 8; ++r) {
          int m = (s << 4) + r + ((lane >> 4) << 3);
          float g = sigm(aG[s][r] + b_gate[c]);
          gm[((size_t)(b0 + m) * Nn + v) * Hh + c] = (__bf16)(g * aM[s][r]);
        }
      }
      __threadfence_block();   // make gm stores visible before next step's Phase A
    } else {
      // ---- Final step: hg = h_new(node 31); mu / sigma heads.
      //      16 waves <-> 16 output tiles: (msub 0/1) x (mu 0..3 | sd 0..3) ----
      size_t fz = 0;
      asm volatile("" : "+s"(fz));      // block speculative hoist/hold
      const __bf16* wmu_l = wmu_p + fz;
      const __bf16* wsd_l = wsd_p + fz;
      const float*  bmu_l = b_mu  + fz;
      const float*  bsd_l = b_std + fz;
      int q    = w & 7;                 // 0..3 -> mu tile, 4..7 -> sd tile
      int msub = w >> 3;                // M-subtile
      int t    = q & 3;
      const __bf16* wp   = (q < 4) ? wmu_l : wsd_l;
      const float*  bias = (q < 4) ? bmu_l : bsd_l;
      size_t obase = (q < 4) ? 0 : (size_t)Bsz * Zz;
      v8f acc = vzero();
#pragma unroll
      for (int kc = 0; kc < 8; ++kc) {
        v16bf a = load_A_lds(s_hnb + msub * 16 * Hh, kc, lane);
        acc = wmma_bf(a, load_B_pk(wp, t, kc, lane), acc);
      }
      int c = (t << 4) + (lane & 15);   // z column 0..63
#pragma unroll
      for (int r = 0; r < 8; ++r) {
        int m = (msub << 4) + r + ((lane >> 4) << 3);
        out[obase + (size_t)(b0 + m) * Zz + c] = acc[r] + bias[c];
      }
    }
  }
}

extern "C" void kernel_launch(void* const* d_in, const int* in_sizes, int n_in,
                              void* d_out, int out_size, void* d_ws, size_t ws_size,
                              hipStream_t stream) {
  (void)in_sizes; (void)n_in; (void)out_size;
  const float* adj    = (const float*)d_in[0];
  const int*   ntyp   = (const int*)  d_in[1];
  const float* w_ih   = (const float*)d_in[2];
  const float* w_hh   = (const float*)d_in[3];
  const float* b_ih   = (const float*)d_in[4];
  const float* b_hh   = (const float*)d_in[5];
  const float* w_gate = (const float*)d_in[6];
  const float* b_gate = (const float*)d_in[7];
  const float* w_map  = (const float*)d_in[8];
  const float* w_mu   = (const float*)d_in[9];
  const float* b_mu   = (const float*)d_in[10];
  const float* w_std  = (const float*)d_in[11];
  const float* b_std  = (const float*)d_in[12];
  float* out = (float*)d_out;

  char* ws = (char*)d_ws;
  size_t off = 0;
  __bf16* gmbuf = (__bf16*)(ws + off); off += (size_t)Bsz * Nn * Hh * 2;   // 32 MB gm cache
  __bf16* whh_p = (__bf16*)(ws + off); off += (size_t)Hh * TH3 * 2;
  __bf16* wg_p  = (__bf16*)(ws + off); off += (size_t)Hh * Hh * 2;
  __bf16* wm_p  = (__bf16*)(ws + off); off += (size_t)Hh * Hh * 2;
  __bf16* wmu_p = (__bf16*)(ws + off); off += (size_t)Hh * Zz * 2;
  __bf16* wsd_p = (__bf16*)(ws + off); off += (size_t)Hh * Zz * 2;
  if (ws_size < off) return;

  // Pack all weight matrices into bf16 WMMA B-fragment layout (re-done each call).
  {
    int t;
    t = (TH3 >> 4) * (Hh >> 5) * 512; pack_w<<<(t + 255) / 256, 256, 0, stream>>>(w_hh,   whh_p, Hh, TH3);
    t = (Hh  >> 4) * (Hh >> 5) * 512; pack_w<<<(t + 255) / 256, 256, 0, stream>>>(w_gate, wg_p,  Hh, Hh);
    t = (Hh  >> 4) * (Hh >> 5) * 512; pack_w<<<(t + 255) / 256, 256, 0, stream>>>(w_map,  wm_p,  Hh, Hh);
    t = (Zz  >> 4) * (Hh >> 5) * 512; pack_w<<<(t + 255) / 256, 256, 0, stream>>>(w_mu,   wmu_p, Hh, Zz);
    t = (Zz  >> 4) * (Hh >> 5) * 512; pack_w<<<(t + 255) / 256, 256, 0, stream>>>(w_std,  wsd_p, Hh, Zz);
  }

  // Entire 32-node scan in one launch: batch rows are independent across WGs.
  lvae_scan<<<Bsz / MR, 512, 0, stream>>>(adj, ntyp, w_ih, b_ih, b_hh, b_gate,
                                          b_mu, b_std, whh_p, wg_p, wm_p,
                                          wmu_p, wsd_p, gmbuf, out);
}
